// GCNModel_48490180772341
// MI455X (gfx1250) — matrix operand order repaired and checked by reading, not compiled, not run on hardware
//
#include <hip/hip_runtime.h>
#include <hip/hip_fp16.h>
#include <math.h>

typedef __attribute__((ext_vector_type(16))) _Float16 v16h;
typedef __attribute__((ext_vector_type(8)))  _Float16 v8h;
typedef __attribute__((ext_vector_type(8)))  float    v8f;

// ---------------------------------------------------------------------------
// One-shot weight repack: W [M,K] fp32 row-major -> f16 packed in the exact
// per-lane B-fragment order this file uses for v_wmma_f32_16x16x32_f16 (contiguous k, not AMD's table order):
//   frag(ft, kc): 32 lanes x 16 halves contiguous;
//   lane L supplies column n = L&15, k = kc*32 + (L>>4)*16 + j.
// One thread produces one lane's 16 halves (4x b128 in, 2x b128 out).
// ---------------------------------------------------------------------------
__global__ void repack_weights_kernel(const float* __restrict__ W,
                                      _Float16* __restrict__ P, int M, int K) {
    const int idx = blockIdx.x * blockDim.x + threadIdx.x;
    const int kcN = K >> 5;
    const int total = (M >> 4) * kcN * 32;
    if (idx >= total) return;
    const int lane = idx & 31;
    const int tmp  = idx >> 5;
    const int kc   = tmp % kcN;
    const int ft   = tmp / kcN;
    const int row  = ft * 16 + (lane & 15);
    const int kb   = kc * 32 + (lane >> 4) * 16;
    const float* src = W + (size_t)row * K + kb;
    _Float16* dst = P + (size_t)idx * 16;
#pragma unroll
    for (int j = 0; j < 16; ++j) dst[j] = (_Float16)src[j];
}

// ---------------------------------------------------------------------------
// Stage a 16-row tile of X (fp32, ld=K) into LDS as f16 (row*K + k),
// optionally with ReLU. Shared by the 8 waves of the block.
// ---------------------------------------------------------------------------
__device__ __forceinline__ void stage_x_tile(const float* __restrict__ X,
                                             int n0, int N, int K, int relu_in,
                                             _Float16* sx) {
    if (n0 + 16 <= N) {
        for (int idx = threadIdx.x; idx < 16 * K; idx += blockDim.x) {
            float v = X[(size_t)n0 * K + idx];
            if (relu_in) v = fmaxf(v, 0.f);
            sx[idx] = (_Float16)v;
        }
    } else {
        for (int idx = threadIdx.x; idx < 16 * K; idx += blockDim.x) {
            const int row = idx / K;
            float v = 0.f;
            if (n0 + row < N) {
                v = X[(size_t)(n0 + row) * K + (idx - row * K)];
                if (relu_in) v = fmaxf(v, 0.f);
            }
            sx[idx] = (_Float16)v;
        }
    }
    __syncthreads();
}

// ---------------------------------------------------------------------------
// One wave: 16(nodes) x 16(features) f32 tile of X @ W^T.
// A from LDS (two aligned v8h loads / chunk), B from packed f16 weights
// (one aligned v16h load / chunk), accumulate with v_wmma_f32_16x16x32_f16.
// ---------------------------------------------------------------------------
__device__ __forceinline__ v8f wmma_xt_wt(const _Float16* __restrict__ sx, int K,
                                          const _Float16* __restrict__ Wp,
                                          int ftb, int lane) {
    const int lo = lane & 15, hi = lane >> 4;
    const int kcN = K >> 5;
    v8f c = {0.f, 0.f, 0.f, 0.f, 0.f, 0.f, 0.f, 0.f};
    const _Float16* bp = Wp + ((size_t)ftb * kcN * 32 + lane) * 16;
    const _Float16* ap = sx + lo * K + hi * 16;
    for (int kc = 0; kc < kcN; ++kc) {
        const v8h a_lo = *(const v8h*)(ap);          // k = kc*32 + hi*16 + 0..7
        const v8h a_hi = *(const v8h*)(ap + 8);      // k = kc*32 + hi*16 + 8..15
        const v16h a = __builtin_shufflevector(a_lo, a_hi,
                0, 1, 2, 3, 4, 5, 6, 7, 8, 9, 10, 11, 12, 13, 14, 15);
        const v16h b = *(const v16h*)(bp);
        c = __builtin_amdgcn_wmma_f32_16x16x32_f16(false, a, false, b,
                                                   (short)0, c, false, false);
        ap += 32;
        bp += 32 * 16;
    }
    return c;
}

// ---------------------------------------------------------------------------
// Prelude: x0 = data @ W_h^T + b_h + pos_encoding(t)   [N,64] -> [N,128]
// ---------------------------------------------------------------------------
__global__ void __launch_bounds__(256)
prelude_kernel(const float* __restrict__ data, const int* __restrict__ t,
               const _Float16* __restrict__ WhP, const float* __restrict__ bh,
               float* __restrict__ x0, int N) {
    __shared__ _Float16 sx[16 * 64];
    const int wave = threadIdx.x >> 5, lane = threadIdx.x & 31;
    const int n0 = blockIdx.x * 16;
    stage_x_tile(data, n0, N, 64, 0, sx);

    v8f c = wmma_xt_wt(sx, 64, WhP, wave, lane);

    const int lo = lane & 15, hi = lane >> 4;
    const int f = wave * 16 + lo;
    // inv_freq = 10000^{-(f & ~1)/128}; cos(x) == sin(x + pi/2) -> branchless
    const float inv  = __expf(-((float)(f & ~1) / 128.f) * 9.210340372f);
    const float phase = (f & 1) ? 1.57079632679f : 0.f;
    const float bias = bh[f];
    float* op = x0 + (size_t)(n0 + hi * 8) * 128 + f;
    if (n0 + 16 <= N) {
#pragma unroll
        for (int r = 0; r < 8; ++r) {
            const float tf = (float)t[n0 + hi * 8 + r];
            op[(size_t)r * 128] = c[r] + bias + __sinf(tf * inv + phase);
        }
    } else {
#pragma unroll
        for (int r = 0; r < 8; ++r) {
            const int node = n0 + hi * 8 + r;
            if (node < N) {
                const float tf = (float)t[node];
                op[(size_t)r * 128] = c[r] + bias + __sinf(tf * inv + phase);
            }
        }
    }
}

// ---------------------------------------------------------------------------
// Fused K/Q/V/S projections of one ResGated layer. S-tile initializes the
// aggregation buffer (agg = X@Ws^T + b). grid=(nodeTiles, (4H/16)/8).
// ---------------------------------------------------------------------------
__global__ void __launch_bounds__(256)
qkvs_kernel(const float* __restrict__ X, int N, int K, int H, int relu_in,
            const _Float16* __restrict__ WkP, const float* __restrict__ bk,
            const _Float16* __restrict__ WqP, const float* __restrict__ bq,
            const _Float16* __restrict__ WvP, const float* __restrict__ bv,
            const _Float16* __restrict__ WsP, const float* __restrict__ bs,
            float* __restrict__ Ko, float* __restrict__ Qo,
            float* __restrict__ Vo, float* __restrict__ Ag) {
    __shared__ _Float16 sx[16 * 256];
    const int wave = threadIdx.x >> 5, lane = threadIdx.x & 31;
    const int n0 = blockIdx.x * 16;
    stage_x_tile(X, n0, N, K, relu_in, sx);

    const int ft = blockIdx.y * 8 + wave;       // tile over 4*H features
    const int f0 = ft * 16;
    const int which = f0 / H, fb = f0 % H;
    const _Float16* Wp;
    const float* bias;
    float* out;
    if (which == 0)      { Wp = WkP; bias = bk; out = Ko; }
    else if (which == 1) { Wp = WqP; bias = bq; out = Qo; }
    else if (which == 2) { Wp = WvP; bias = bv; out = Vo; }
    else                 { Wp = WsP; bias = bs; out = Ag; }

    v8f c = wmma_xt_wt(sx, K, Wp, fb >> 4, lane);

    const int lo = lane & 15, hi = lane >> 4;
    const int f = fb + lo;
    const float bval = bias[f];
    float* op = out + (size_t)(n0 + hi * 8) * H + f;
    if (n0 + 16 <= N) {
#pragma unroll
        for (int r = 0; r < 8; ++r) op[(size_t)r * H] = c[r] + bval;
    } else {
#pragma unroll
        for (int r = 0; r < 8; ++r)
            if (n0 + hi * 8 + r < N) op[(size_t)r * H] = c[r] + bval;
    }
}

// ---------------------------------------------------------------------------
// Edge phase: agg[dst] += sigmoid(k[dst] + q[src]) * v[src]
// One wave32 per edge; float4 gathers (rows are L2-resident); hardware fp32
// atomics; global_prefetch_b8 on the scatter line.
// ---------------------------------------------------------------------------
__global__ void __launch_bounds__(256)
edge_kernel(const float* __restrict__ Kb, const float* __restrict__ Qb,
            const float* __restrict__ Vb, const int* __restrict__ src,
            const int* __restrict__ dst, float* __restrict__ Agg,
            int E, int C) {
    const int wave = threadIdx.x >> 5, lane = threadIdx.x & 31;
    const int e = blockIdx.x * (blockDim.x >> 5) + wave;
    if (e >= E) return;
    const int s = src[e], d = dst[e];
    const float4* kd = (const float4*)(Kb + (size_t)d * C);
    const float4* qs = (const float4*)(Qb + (size_t)s * C);
    const float4* vs = (const float4*)(Vb + (size_t)s * C);
    float* ag = Agg + (size_t)d * C;
    __builtin_prefetch(ag + lane * 4, 1, 0);

    for (int c4 = lane; c4 * 4 < C; c4 += 32) {
        const float4 kv = kd[c4], qv = qs[c4], vv = vs[c4];
        float4 m;
        m.x = vv.x / (1.f + __expf(-(kv.x + qv.x)));
        m.y = vv.y / (1.f + __expf(-(kv.y + qv.y)));
        m.z = vv.z / (1.f + __expf(-(kv.z + qv.z)));
        m.w = vv.w / (1.f + __expf(-(kv.w + qv.w)));
        const int cb = c4 * 4;
        __hip_atomic_fetch_add(&ag[cb + 0], m.x, __ATOMIC_RELAXED, __HIP_MEMORY_SCOPE_AGENT);
        __hip_atomic_fetch_add(&ag[cb + 1], m.y, __ATOMIC_RELAXED, __HIP_MEMORY_SCOPE_AGENT);
        __hip_atomic_fetch_add(&ag[cb + 2], m.z, __ATOMIC_RELAXED, __HIP_MEMORY_SCOPE_AGENT);
        __hip_atomic_fetch_add(&ag[cb + 3], m.w, __ATOMIC_RELAXED, __HIP_MEMORY_SCOPE_AGENT);
    }
}

// ---------------------------------------------------------------------------
static inline void repack(const float* W, _Float16* P, int M, int K,
                          hipStream_t stream) {
    const int total = (M >> 4) * (K >> 5) * 32;
    repack_weights_kernel<<<(total + 255) / 256, 256, 0, stream>>>(W, P, M, K);
}

extern "C" void kernel_launch(void* const* d_in, const int* in_sizes, int n_in,
                              void* d_out, int out_size, void* d_ws, size_t ws_size,
                              hipStream_t stream) {
    (void)n_in; (void)out_size; (void)ws_size;
    const float* data = (const float*)d_in[0];
    const int*   ei   = (const int*)d_in[1];     // [2,E] flat: src then dst
    const int*   t    = (const int*)d_in[2];
    const float* Wh   = (const float*)d_in[3];
    const float* bh   = (const float*)d_in[4];
    const float* Wk1  = (const float*)d_in[5];  const float* bk1 = (const float*)d_in[6];
    const float* Wq1  = (const float*)d_in[7];  const float* bq1 = (const float*)d_in[8];
    const float* Wv1  = (const float*)d_in[9];  const float* bv1 = (const float*)d_in[10];
    const float* Ws1  = (const float*)d_in[11]; const float* b1  = (const float*)d_in[12];
    const float* Wk2  = (const float*)d_in[13]; const float* bk2 = (const float*)d_in[14];
    const float* Wq2  = (const float*)d_in[15]; const float* bq2 = (const float*)d_in[16];
    const float* Wv2  = (const float*)d_in[17]; const float* bv2 = (const float*)d_in[18];
    const float* Ws2  = (const float*)d_in[19]; const float* b2  = (const float*)d_in[20];

    const int N = in_sizes[0] / 64;     // 50000
    const int E = in_sizes[1] / 2;      // 800000
    const int TIME = 128, HID = 256, OUT = 128;
    const int* src = ei;
    const int* dst = ei + E;

    // workspace layout: fp32 activations first, then f16 packed weights
    float* ws = (float*)d_ws;
    float* x0 = ws;                          // N*128
    float* k1 = x0 + (size_t)N * TIME;       // N*256 each
    float* q1 = k1 + (size_t)N * HID;
    float* v1 = q1 + (size_t)N * HID;
    float* h1 = v1 + (size_t)N * HID;        // conv1 aggregation
    float* k2 = h1 + (size_t)N * HID;        // N*128 each
    float* q2 = k2 + (size_t)N * OUT;
    float* v2 = q2 + (size_t)N * OUT;
    float* fend = v2 + (size_t)N * OUT;

    _Float16* hp = (_Float16*)fend;
    _Float16* pWh  = hp;                 hp += (size_t)TIME * 64;   // 128x64
    _Float16* pWk1 = hp;                 hp += (size_t)HID * TIME;  // 256x128 each
    _Float16* pWq1 = hp;                 hp += (size_t)HID * TIME;
    _Float16* pWv1 = hp;                 hp += (size_t)HID * TIME;
    _Float16* pWs1 = hp;                 hp += (size_t)HID * TIME;
    _Float16* pWk2 = hp;                 hp += (size_t)OUT * HID;   // 128x256 each
    _Float16* pWq2 = hp;                 hp += (size_t)OUT * HID;
    _Float16* pWv2 = hp;                 hp += (size_t)OUT * HID;
    _Float16* pWs2 = hp;

    float* outp = (float*)d_out;             // conv2 aggregation == final output

    // 0) repack all weights to f16 B-fragment order (tiny, L2-resident)
    repack(Wh,  pWh,  TIME, 64,   stream);
    repack(Wk1, pWk1, HID,  TIME, stream);
    repack(Wq1, pWq1, HID,  TIME, stream);
    repack(Wv1, pWv1, HID,  TIME, stream);
    repack(Ws1, pWs1, HID,  TIME, stream);
    repack(Wk2, pWk2, OUT,  HID,  stream);
    repack(Wq2, pWq2, OUT,  HID,  stream);
    repack(Wv2, pWv2, OUT,  HID,  stream);
    repack(Ws2, pWs2, OUT,  HID,  stream);

    const int nodeTiles = (N + 15) / 16;
    const int edgeBlocks = (E + 7) / 8;

    // 1) x0 = data @ Wh^T + bh + pos_encoding(t)
    prelude_kernel<<<dim3(nodeTiles), 256, 0, stream>>>(data, t, pWh, bh, x0, N);

    // 2) conv1 projections; h1 initialized with x0 @ Ws1^T + b1
    qkvs_kernel<<<dim3(nodeTiles, HID / 32), 256, 0, stream>>>(
        x0, N, TIME, HID, 0,
        pWk1, bk1, pWq1, bq1, pWv1, bv1, pWs1, b1, k1, q1, v1, h1);

    // 3) conv1 edge aggregation into h1
    edge_kernel<<<dim3(edgeBlocks), 256, 0, stream>>>(k1, q1, v1, src, dst, h1, E, HID);

    // 4) conv2 projections on relu(h1); d_out initialized with relu(h1) @ Ws2^T + b2
    qkvs_kernel<<<dim3(nodeTiles, OUT / 32), 256, 0, stream>>>(
        h1, N, HID, OUT, 1,
        pWk2, bk2, pWq2, bq2, pWv2, bv2, pWs2, b2, k2, q2, v2, outp);

    // 5) conv2 edge aggregation into d_out
    edge_kernel<<<dim3(edgeBlocks), 256, 0, stream>>>(k2, q2, v2, src, dst, outp, E, OUT);
}
